// DVE_loss_multi_47055661695057
// MI455X (gfx1250) — matrix operand order repaired and checked by reading, not compiled, not run on hardware
//
#include <hip/hip_runtime.h>

// ---------------------------------------------------------------------------
// DVE loss for MI455X (gfx1250, wave32). Attention and GEMM contractions use
// v_wmma_f32_16x16x32_bf16. Sinkhorn kept L2-resident (32MB << 192MB L2).
// Workspace use: ~103 MB.
// ---------------------------------------------------------------------------

typedef __bf16 bf16_t;
typedef __attribute__((ext_vector_type(16))) __bf16 v16bf;
typedef __attribute__((ext_vector_type(8)))  __bf16 v8bf;
typedef __attribute__((ext_vector_type(8)))  float  v8f;

#define Bz   8
#define Nn   1024
#define Cc   64
#define NEGINF (-3.0e38f)

__device__ __forceinline__ bf16_t f2bf(float f) {
  unsigned int u = __float_as_uint(f);
  unsigned int r = (u + 0x7FFFu + ((u >> 16) & 1u)) >> 16;
  unsigned short s = (unsigned short)r;
  return __builtin_bit_cast(__bf16, s);
}

__device__ __forceinline__ v8f wmma_bf16(const v16bf& a, const v16bf& b, const v8f& c) {
  return __builtin_amdgcn_wmma_f32_16x16x32_bf16(false, a, false, b, (short)0, c, false, false);
}

// -------------------- stage 0: f32 -> bf16 (+ transposed copy) --------------
__global__ __launch_bounds__(256)
void convert_kernel(const float* __restrict__ feats,
                    bf16_t* __restrict__ f1h, bf16_t* __restrict__ f2h,
                    bf16_t* __restrict__ f1t) {
  size_t i = (size_t)blockIdx.x * 256 + threadIdx.x;     // over 8*1024*64
  int c = (int)(i & 63);
  size_t n = (i >> 6) & 1023;
  size_t b = i >> 16;
  float v1 = feats[(((size_t)(2*b)  ) * Nn + n) * Cc + c];
  float v2 = feats[(((size_t)(2*b)+1) * Nn + n) * Cc + c];
  f1h[i] = f2bf(v1);
  f2h[i] = f2bf(v2);
  f1t[(b * Cc + c) * Nn + n] = f2bf(v1);
}

// -------------------- fused attention (flash, two-pass, WMMA) ---------------
// out = softmax(Q · K^T) · V  with Q rows = f1h[b], keys = f1h[kb] (kb loops
// over nj batches), values read channel-major from f1t[kb].
__global__ __launch_bounds__(32)
void attn_kernel(const bf16_t* __restrict__ qrm,   // [8][1024][64]
                 const bf16_t* __restrict__ krm,   // [8][1024][64]
                 const bf16_t* __restrict__ vt,    // [8][64][1024]
                 bf16_t* __restrict__ out_bf,      // [8][1024][64] or null
                 float*  __restrict__ out_f,       // [8][1024][64] or null
                 int nj, int selfmode) {
  const int b    = blockIdx.y;
  const int row0 = blockIdx.x * 16;
  const int lane = threadIdx.x;
  const int lh   = lane >> 4;        // half-wave selector
  const int l16  = lane & 15;

  __shared__ __align__(32) float  s_red[16 * 16];
  __shared__               float  s_rowmax[16];
  __shared__               float  s_rowsum[16];
  __shared__ __align__(32) bf16_t s_p[16 * 32];

  // Q A-operand, both 32-wide K chunks (C=64)
  v16bf qa[2];
  const bf16_t* qbase = qrm + ((size_t)b * Nn + row0 + l16) * Cc;
  for (int kci = 0; kci < 2; ++kci) {
    int kc = kci * 32;
    v8bf lo = *(const v8bf*)(qbase + kc + lh * 16);
    v8bf hi = *(const v8bf*)(qbase + kc + lh * 16 + 8);
    for (int i = 0; i < 8; ++i) { qa[kci][i] = lo[i]; qa[kci][i + 8] = hi[i]; }
  }

  // ---- pass A: row maxima of all logits ----
  float rmax8[8];
  for (int r = 0; r < 8; ++r) rmax8[r] = NEGINF;
  for (int j = 0; j < nj; ++j) {
    int kb = selfmode ? b : ((b + 1 + j) & 7);
    const bf16_t* kbase = krm + (size_t)kb * Nn * Cc;
    for (int mc = 0; mc < Nn; mc += 32) {
      for (int nt = 0; nt < 2; ++nt) {
        v8f s; for (int i = 0; i < 8; ++i) s[i] = 0.f;
        for (int kci = 0; kci < 2; ++kci) {
          int kc = kci * 32;
          v16bf bm = *(const v16bf*)(kbase + (size_t)(mc + nt * 16 + l16) * Cc + kc + lh * 16);
          s = wmma_bf16(qa[kci], bm, s);
        }
        for (int r = 0; r < 8; ++r) rmax8[r] = fmaxf(rmax8[r], s[r]);
      }
    }
  }
  for (int r = 0; r < 8; ++r) s_red[(r + 8 * lh) * 16 + l16] = rmax8[r];
  __syncthreads();
  if (lane < 16) {
    float m = NEGINF;
    for (int i = 0; i < 16; ++i) m = fmaxf(m, s_red[lane * 16 + i]);
    s_rowmax[lane] = m;
  }
  __syncthreads();
  float myrowmax[8];
  for (int r = 0; r < 8; ++r) myrowmax[r] = s_rowmax[r + 8 * lh];

  // ---- pass B: P = exp(S - max) (unnormalized), O += P·V ----
  float sum8[8]; for (int r = 0; r < 8; ++r) sum8[r] = 0.f;
  v8f oacc[4];
  for (int t = 0; t < 4; ++t) for (int i = 0; i < 8; ++i) oacc[t][i] = 0.f;

  for (int j = 0; j < nj; ++j) {
    int kb = selfmode ? b : ((b + 1 + j) & 7);
    const bf16_t* kbase = krm + (size_t)kb * Nn * Cc;
    const bf16_t* vbase = vt  + (size_t)kb * Cc * Nn;
    for (int mc = 0; mc < Nn; mc += 32) {
      for (int nt = 0; nt < 2; ++nt) {
        v8f s; for (int i = 0; i < 8; ++i) s[i] = 0.f;
        for (int kci = 0; kci < 2; ++kci) {
          int kc = kci * 32;
          v16bf bm = *(const v16bf*)(kbase + (size_t)(mc + nt * 16 + l16) * Cc + kc + lh * 16);
          s = wmma_bf16(qa[kci], bm, s);
        }
        for (int r = 0; r < 8; ++r) {
          float e = __expf(s[r] - myrowmax[r]);
          sum8[r] += e;
          s_p[(r + 8 * lh) * 32 + nt * 16 + l16] = f2bf(e);
        }
      }
      __syncthreads();
      // reload P as A-operand (16x32)
      v16bf pa;
      {
        const bf16_t* pb = &s_p[l16 * 32];
        v8bf lo = *(const v8bf*)(pb + lh * 16);
        v8bf hi = *(const v8bf*)(pb + lh * 16 + 8);
        for (int i = 0; i < 8; ++i) { pa[i] = lo[i]; pa[i + 8] = hi[i]; }
      }
      for (int ct = 0; ct < 4; ++ct) {
        v16bf bv = *(const v16bf*)(vbase + (size_t)(ct * 16 + l16) * Nn + mc + lh * 16);
        oacc[ct] = wmma_bf16(pa, bv, oacc[ct]);
      }
      __syncthreads();
    }
  }

  // row sums -> normalize -> store
  for (int r = 0; r < 8; ++r) s_red[(r + 8 * lh) * 16 + l16] = sum8[r];
  __syncthreads();
  if (lane < 16) {
    float sm = 0.f;
    for (int i = 0; i < 16; ++i) sm += s_red[lane * 16 + i];
    s_rowsum[lane] = sm;
  }
  __syncthreads();
  float inv8[8];
  for (int r = 0; r < 8; ++r) inv8[r] = 1.0f / s_rowsum[r + 8 * lh];

  for (int ct = 0; ct < 4; ++ct)
    for (int r = 0; r < 8; ++r) {
      float o = oacc[ct][r] * inv8[r];
      size_t oi = ((size_t)b * Nn + row0 + r + 8 * lh) * Cc + ct * 16 + l16;
      if (out_bf) out_bf[oi] = f2bf(o);
      if (out_f)  out_f[oi]  = o;
    }
}

// -------------------- GEMM: C[b] = A[b] · Bm[b]^T (WMMA) --------------------
__global__ __launch_bounds__(32)
void gemm_nt_kernel(const bf16_t* __restrict__ A,   // [8][1024][64]
                    const bf16_t* __restrict__ Bm,  // [8][1024][64]
                    float* __restrict__ C) {        // [8][1024][1024]
  const int b = blockIdx.z, row0 = blockIdx.x * 16, col0 = blockIdx.y * 64;
  const int lane = threadIdx.x, lh = lane >> 4, l16 = lane & 15;

  v16bf qa[2];
  const bf16_t* abase = A + ((size_t)b * Nn + row0 + l16) * Cc;
  for (int kci = 0; kci < 2; ++kci) {
    int kc = kci * 32;
    v8bf lo = *(const v8bf*)(abase + kc + lh * 16);
    v8bf hi = *(const v8bf*)(abase + kc + lh * 16 + 8);
    for (int i = 0; i < 8; ++i) { qa[kci][i] = lo[i]; qa[kci][i + 8] = hi[i]; }
  }
  const bf16_t* bbase = Bm + (size_t)b * Nn * Cc;
  for (int ct = 0; ct < 4; ++ct) {
    v8f acc; for (int i = 0; i < 8; ++i) acc[i] = 0.f;
    for (int kci = 0; kci < 2; ++kci) {
      int kc = kci * 32;
      v16bf bm = *(const v16bf*)(bbase + (size_t)(col0 + ct * 16 + l16) * Cc + kc + lh * 16);
      acc = wmma_bf16(qa[kci], bm, acc);
    }
    for (int r = 0; r < 8; ++r)
      C[((size_t)b * Nn + row0 + r + 8 * lh) * Nn + col0 + ct * 16 + l16] = acc[r];
  }
}

// -------------------- row softmax stats -------------------------------------
__global__ __launch_bounds__(256)
void rowstats_kernel(const float* __restrict__ Mat, float* __restrict__ rmax,
                     float* __restrict__ rsum) {
  const int b = blockIdx.y, r = blockIdx.x, tid = threadIdx.x;
  size_t base = ((size_t)b * Nn + r) * Nn;
  __shared__ float red[256];
  float v[4], mx = NEGINF;
  for (int k = 0; k < 4; ++k) { v[k] = Mat[base + tid + k * 256]; mx = fmaxf(mx, v[k]); }
  red[tid] = mx; __syncthreads();
  for (int s = 128; s > 0; s >>= 1) { if (tid < s) red[tid] = fmaxf(red[tid], red[tid + s]); __syncthreads(); }
  mx = red[0]; __syncthreads();
  float sm = 0.f;
  for (int k = 0; k < 4; ++k) sm += __expf(v[k] - mx);
  red[tid] = sm; __syncthreads();
  for (int s = 128; s > 0; s >>= 1) { if (tid < s) red[tid] += red[tid + s]; __syncthreads(); }
  if (tid == 0) { rmax[(size_t)b * Nn + r] = mx; rsum[(size_t)b * Nn + r] = red[0]; }
}

// -------------------- sinkhorn ----------------------------------------------
__global__ __launch_bounds__(256)
void sink_init_kernel(const float* __restrict__ corrA, float* __restrict__ la) {
  size_t i = (size_t)blockIdx.x * 256 + threadIdx.x;
  la[i] = corrA[i] * (1.0f / 0.7f);
}

__global__ __launch_bounds__(256)
void sink_row_kernel(float* __restrict__ la) {
  const int b = blockIdx.y, r = blockIdx.x, tid = threadIdx.x;
  size_t base = ((size_t)b * Nn + r) * Nn;
  __shared__ float red[256];
  float v[4], mx = NEGINF;
  for (int k = 0; k < 4; ++k) { v[k] = la[base + tid + k * 256]; mx = fmaxf(mx, v[k]); }
  red[tid] = mx; __syncthreads();
  for (int s = 128; s > 0; s >>= 1) { if (tid < s) red[tid] = fmaxf(red[tid], red[tid + s]); __syncthreads(); }
  mx = red[0]; __syncthreads();
  float sm = 0.f;
  for (int k = 0; k < 4; ++k) sm += __expf(v[k] - mx);
  red[tid] = sm; __syncthreads();
  for (int s = 128; s > 0; s >>= 1) { if (tid < s) red[tid] += red[tid + s]; __syncthreads(); }
  float lse = mx + __logf(red[0]);
  for (int k = 0; k < 4; ++k) la[base + tid + k * 256] = v[k] - lse;
}

__global__ __launch_bounds__(256)
void sink_col_kernel(float* __restrict__ la) {
  const int b = blockIdx.y;
  const int col = blockIdx.x * 256 + threadIdx.x;
  size_t base = (size_t)b * Nn * Nn + col;
  float mx = NEGINF;
  for (int r = 0; r < Nn; ++r) mx = fmaxf(mx, la[base + (size_t)r * Nn]);
  float sm = 0.f;
  for (int r = 0; r < Nn; ++r) sm += __expf(la[base + (size_t)r * Nn] - mx);
  float lse = mx + __logf(sm);
  for (int r = 0; r < Nn; ++r) la[base + (size_t)r * Nn] -= lse;
}

// -------------------- main loss accumulation --------------------------------
__global__ __launch_bounds__(256)
void loss_kernel(const float* __restrict__ corrA, const float* __restrict__ corr12,
                 const float* __restrict__ la,
                 const float* __restrict__ rmaxA, const float* __restrict__ rsumA,
                 const float* __restrict__ rmax12, const float* __restrict__ rsum12,
                 const float* __restrict__ pc0, float* __restrict__ acc) {
  const int b = blockIdx.y, r = blockIdx.x, tid = threadIdx.x;
  size_t rowi = (size_t)b * Nn + r;
  size_t base = rowi * Nn;
  float pX = pc0[rowi * 3 + 0], pY = pc0[rowi * 3 + 1], pZ = pc0[rowi * 3 + 2];
  float rmA = rmaxA[rowi],  rsA  = 1.0f / rsumA[rowi];
  float rm2 = rmax12[rowi], rs2  = 1.0f / rsum12[rowi];
  float lossAcc = 0.f, lcAcc = 0.f;
  float bestv = NEGINF; int bestm = 0;
  for (int k = 0; k < 4; ++k) {
    int m = tid + k * 256;
    float vA   = corrA[base + m];
    float smA  = __expf(vA - rmA) * rsA;
    float sm12 = __expf(corr12[base + m] - rm2) * rs2;
    float sink = __expf(la[base + m]);
    size_t mi = ((size_t)b * Nn + m) * 3;
    float dx = pX - pc0[mi], dy = pY - pc0[mi + 1], dz = pZ - pc0[mi + 2];
    float diff = sqrtf(sqrtf(dx * dx + dy * dy + dz * dz));
    lossAcc += diff * (smA + 0.5f * sm12);
    lcAcc   += fabsf(sink - smA);
    if (vA > bestv || (vA == bestv && m < bestm)) { bestv = vA; bestm = m; }
  }
  __shared__ float redA[256], redB[256], redV[256];
  __shared__ int   redM[256];
  redA[tid] = lossAcc; redB[tid] = lcAcc; redV[tid] = bestv; redM[tid] = bestm;
  __syncthreads();
  for (int s = 128; s > 0; s >>= 1) {
    if (tid < s) {
      redA[tid] += redA[tid + s];
      redB[tid] += redB[tid + s];
      if (redV[tid + s] > redV[tid] ||
          (redV[tid + s] == redV[tid] && redM[tid + s] < redM[tid])) {
        redV[tid] = redV[tid + s]; redM[tid] = redM[tid + s];
      }
    }
    __syncthreads();
  }
  if (tid == 0) {
    atomicAdd(&acc[0], redA[0]);
    atomicAdd(&acc[1], redB[0]);
    if (redM[0] == r) atomicAdd(&acc[2], 1.0f);
  }
}

// -------------------- recon diagonal term -----------------------------------
__global__ __launch_bounds__(256)
void recon_kernel(const float* __restrict__ fv, const float* __restrict__ feats,
                  float* __restrict__ acc) {
  const int b = blockIdx.y, r = blockIdx.x, tid = threadIdx.x;
  __shared__ float qrow[64];
  __shared__ float dots[1024];
  __shared__ float red[256];
  if (tid < 64) qrow[tid] = fv[((size_t)b * Nn + r) * Cc + tid];
  __syncthreads();
  const float* f1 = feats + (size_t)(2 * b) * Nn * Cc;
  for (int k = 0; k < 4; ++k) {
    int m = tid + k * 256;
    const float* row = f1 + (size_t)m * Cc;
    float d = 0.f;
    for (int c = 0; c < Cc; ++c) d += qrow[c] * row[c];
    dots[m] = d;
  }
  __syncthreads();
  float mx = NEGINF;
  for (int k = 0; k < 4; ++k) mx = fmaxf(mx, dots[tid + k * 256]);
  red[tid] = mx; __syncthreads();
  for (int s = 128; s > 0; s >>= 1) { if (tid < s) red[tid] = fmaxf(red[tid], red[tid + s]); __syncthreads(); }
  mx = red[0]; __syncthreads();
  float sm = 0.f;
  for (int k = 0; k < 4; ++k) sm += __expf(dots[tid + k * 256] - mx);
  red[tid] = sm; __syncthreads();
  for (int s = 128; s > 0; s >>= 1) { if (tid < s) red[tid] += red[tid + s]; __syncthreads(); }
  if (tid == 0) atomicAdd(acc, __expf(dots[r] - mx) / red[0]);
}

// -------------------- tiny kernels ------------------------------------------
__global__ void zero_kernel(float* acc) { if (threadIdx.x < 8) acc[threadIdx.x] = 0.f; }

__global__ void finalize_kernel(const float* __restrict__ acc, float* __restrict__ out) {
  float lossSum = acc[0], lcSum = acc[1], match = acc[2], recon = acc[3];
  const float Nf = (float)Nn, Bf = (float)Bz;
  float loss  = lossSum / Nf;
  float Lc    = 3.0f * lcSum / Nf;
  float total = loss + 0.01f * Lc;
  out[0] = total / Bf;
  out[1] = loss / Bf;
  out[2] = Lc / Bf;
  out[3] = match / Bf;
  out[4] = -recon / (Nf * Bf);
}

// ---------------------------------------------------------------------------
extern "C" void kernel_launch(void* const* d_in, const int* in_sizes, int n_in,
                              void* d_out, int out_size, void* d_ws, size_t ws_size,
                              hipStream_t stream) {
  const float* feats = (const float*)d_in[0];   // (16,1024,64) f32
  const float* pc0   = (const float*)d_in[1];   // (8,1024,3)   f32
  float* out = (float*)d_out;                   // 5 scalars

  char* ws = (char*)d_ws;
  size_t off = 0;
  auto take = [&](size_t bytes) -> char* {
    char* p = ws + off;
    off = (off + bytes + 255) & ~(size_t)255;
    return p;
  };
  const size_t NF = (size_t)Bz * Nn * Cc;       // 512K elems
  const size_t NM = (size_t)Bz * Nn * Nn;       // 8M elems
  bf16_t* f1h   = (bf16_t*)take(NF * 2);
  bf16_t* f2h   = (bf16_t*)take(NF * 2);
  bf16_t* f1t   = (bf16_t*)take(NF * 2);
  bf16_t* fvh   = (bf16_t*)take(NF * 2);        // f1_via_fa (bf16)
  float*  fvf   = (float*) take(NF * 4);        // f1v (f32)
  float*  corrA = (float*) take(NM * 4);        // corr_1a2
  float*  corr12= (float*) take(NM * 4);
  float*  la    = (float*) take(NM * 4);        // sinkhorn log matrix
  float*  rmaxA = (float*) take((size_t)Bz * Nn * 4);
  float*  rsumA = (float*) take((size_t)Bz * Nn * 4);
  float*  rmax12= (float*) take((size_t)Bz * Nn * 4);
  float*  rsum12= (float*) take((size_t)Bz * Nn * 4);
  float*  acc   = (float*) take(8 * 4);

  zero_kernel<<<1, 32, 0, stream>>>(acc);
  convert_kernel<<<(unsigned)(NF / 256), 256, 0, stream>>>(feats, f1h, f2h, f1t);

  // attention chain 1: keys/values = 3 rotated batches -> f1_via_fa (bf16)
  attn_kernel<<<dim3(64, 8), 32, 0, stream>>>(f1h, f1h, f1t, fvh, nullptr, 3, 0);
  // self-attention chain: -> f1v (f32)
  attn_kernel<<<dim3(64, 8), 32, 0, stream>>>(f1h, f1h, f1t, nullptr, fvf, 1, 1);

  gemm_nt_kernel<<<dim3(64, 16, 8), 32, 0, stream>>>(fvh, f2h, corrA);   // corr_1a2
  gemm_nt_kernel<<<dim3(64, 16, 8), 32, 0, stream>>>(f1h, f2h, corr12);  // corr_12

  rowstats_kernel<<<dim3(1024, 8), 256, 0, stream>>>(corrA,  rmaxA,  rsumA);
  rowstats_kernel<<<dim3(1024, 8), 256, 0, stream>>>(corr12, rmax12, rsum12);

  sink_init_kernel<<<(unsigned)(NM / 256), 256, 0, stream>>>(corrA, la);
  for (int it = 0; it < 20; ++it) {
    sink_row_kernel<<<dim3(1024, 8), 256, 0, stream>>>(la);
    sink_col_kernel<<<dim3(4, 8),    256, 0, stream>>>(la);
  }

  loss_kernel<<<dim3(1024, 8), 256, 0, stream>>>(corrA, corr12, la,
                                                 rmaxA, rsumA, rmax12, rsum12,
                                                 pc0, acc);
  recon_kernel<<<dim3(1024, 8), 256, 0, stream>>>(fvf, feats, &acc[3]);
  finalize_kernel<<<1, 1, 0, stream>>>(acc, out);
}
